// RWKV7_TimeMix_57320633532864
// MI455X (gfx1250) — matrix operand order repaired and checked by reading, not compiled, not run on hardware
//
#include <hip/hip_runtime.h>
#include <hip/hip_bf16.h>

typedef __bf16 bf16;
typedef __attribute__((ext_vector_type(16))) __bf16 v16bf;
typedef __attribute__((ext_vector_type(8)))  __bf16 v8bf;
typedef __attribute__((ext_vector_type(8)))  float  v8f;

#define DEV __device__ __forceinline__

static constexpr int Bsz = 4, Tsz = 1024, Csz = 1024, Hn = 16, Nn = 64;
static constexpr int Msz = Bsz * Tsz;              // 4096 rows
static constexpr size_t MC = (size_t)Msz * Csz;    // 4,194,304 elements
static constexpr size_t CC = (size_t)Csz * Csz;

DEV float sigf(float x) { return 1.0f / (1.0f + expf(-x)); }

DEV float wave_sum32(float v) {
#pragma unroll
    for (int off = 16; off > 0; off >>= 1) v += __shfl_xor(v, off, 32);
    return v;
}

// ---------------------------------------------------------------- fp32 -> bf16
__global__ void cvt_bf16_kernel(const float* __restrict__ src, bf16* __restrict__ dst, int n) {
    int i = blockIdx.x * blockDim.x + threadIdx.x;
    if (i < n) dst[i] = (bf16)src[i];
}

// ------------------------------------------------- token shift + 6-way mixing
__global__ void prep_mix_kernel(const float* __restrict__ x,
                                const float* __restrict__ mr, const float* __restrict__ mw,
                                const float* __restrict__ mk, const float* __restrict__ mv,
                                const float* __restrict__ ma, const float* __restrict__ mg,
                                bf16* __restrict__ oxr, bf16* __restrict__ oxw,
                                bf16* __restrict__ oxk, bf16* __restrict__ oxv,
                                bf16* __restrict__ oxa, bf16* __restrict__ oxg) {
    int idx = blockIdx.x * blockDim.x + threadIdx.x;
    if (idx >= (int)MC) return;
    int c = idx & (Csz - 1);
    int t = (idx >> 10) & (Tsz - 1);
    float xv = x[idx];
    float xp = (t == 0) ? 0.0f : x[idx - Csz];
    float dx = xp - xv;  // shifted - x
    oxr[idx] = (bf16)(xv + dx * mr[c]);
    oxw[idx] = (bf16)(xv + dx * mw[c]);
    oxk[idx] = (bf16)(xv + dx * mk[c]);
    oxv[idx] = (bf16)(xv + dx * mv[c]);
    oxa[idx] = (bf16)(xv + dx * ma[c]);
    oxg[idx] = (bf16)(xv + dx * mg[c]);
}

// ------------------------------------------------------------ bf16 WMMA GEMM
// Y[M,N] = A[M,K] @ W[N,K]^T   (Y[m,n] = sum_k A[m,k]*W[n,k])
// Register blocking: one wave computes a 16x64 output strip (4 accumulators),
// reusing one A fragment across 4 WMMAs per k-step. The 8 waves of a block
// share the same N-range (wave selects the M-tile), so their B-fragment
// addresses coincide -> B is fetched once per block via WGP$/L2.
__global__ void gemm_bf16_wmma_kernel(const bf16* __restrict__ A,
                                      const bf16* __restrict__ W,
                                      float* __restrict__ Y,
                                      int M, int N, int K) {
    const int lane = threadIdx.x & 31;
    const int wave = threadIdx.x >> 5;
    const int m0 = blockIdx.y * 128 + wave * 16;   // per-wave M tile
    const int n0 = blockIdx.x * 64;                // block-wide N strip (4 x 16)
    const int half = lane >> 4;                    // K-half selector
    const int lr   = lane & 15;                    // row-in-tile (A: M, B: N)

    const bf16* arow = A + (size_t)(m0 + lr) * K;
    const bf16* brow0 = W + (size_t)(n0 +  0 + lr) * K;
    const bf16* brow1 = W + (size_t)(n0 + 16 + lr) * K;
    const bf16* brow2 = W + (size_t)(n0 + 32 + lr) * K;
    const bf16* brow3 = W + (size_t)(n0 + 48 + lr) * K;

    v8f acc0 = {}, acc1 = {}, acc2 = {}, acc3 = {};
    for (int k = 0; k < K; k += 32) {
        // A 16x32 bf16 fragment: lane<16 holds K 0..15, lane>=16 holds K 16..31
        v16bf afrag = *(const v16bf*)(arow + k + half * 16);
        // B 32x16 bf16 fragments: lane<16 holds K 0..15, lane>=16 holds K 16..31
        v16bf b0 = *(const v16bf*)(brow0 + k + half * 16);
        v16bf b1 = *(const v16bf*)(brow1 + k + half * 16);
        v16bf b2 = *(const v16bf*)(brow2 + k + half * 16);
        v16bf b3 = *(const v16bf*)(brow3 + k + half * 16);
        acc0 = __builtin_amdgcn_wmma_f32_16x16x32_bf16(false, afrag, false, b0,
                                                       (short)0, acc0, false, false);
        acc1 = __builtin_amdgcn_wmma_f32_16x16x32_bf16(false, afrag, false, b1,
                                                       (short)0, acc1, false, false);
        acc2 = __builtin_amdgcn_wmma_f32_16x16x32_bf16(false, afrag, false, b2,
                                                       (short)0, acc2, false, false);
        acc3 = __builtin_amdgcn_wmma_f32_16x16x32_bf16(false, afrag, false, b3,
                                                       (short)0, acc3, false, false);
    }
    // D 16x16 f32: lane%16 = N col; VGPR v = row v + 8*(lane/16)
#pragma unroll
    for (int v = 0; v < 8; ++v) {
        size_t row = (size_t)(m0 + v + 8 * half) * N;
        Y[row + n0 +  0 + lr] = acc0[v];
        Y[row + n0 + 16 + lr] = acc1[v];
        Y[row + n0 + 32 + lr] = acc2[v];
        Y[row + n0 + 48 + lr] = acc3[v];
    }
}

// --------------------------------------------------------------- LoRA stage 1
// H[m,d] = act( sum_k X[m,k] * U[k,d] );  blockDim.x == D (64 or 160)
__global__ void lora_stage1_kernel(const bf16* __restrict__ X, const float* __restrict__ U,
                                   float* __restrict__ Hout, int K, int D, int act) {
    int m = blockIdx.x, d = threadIdx.x;
    const bf16* xr = X + (size_t)m * K;
    float acc = 0.0f;
    for (int k = 0; k < K; ++k) acc += (float)xr[k] * U[(size_t)k * D + d];
    if (act == 1) acc = tanhf(acc);
    else if (act == 2) acc = sigf(acc);
    Hout[(size_t)m * D + d] = acc;
}

// ----------------------------------------------- LoRA stage 2 variants (per m)
__global__ void lora_w_kernel(const float* __restrict__ Hw, const float* __restrict__ w2,
                              const float* __restrict__ w0, float* __restrict__ decay) {
    int m = blockIdx.x, tid = threadIdx.x;
    __shared__ float hs[64];
    if (tid < 64) hs[tid] = Hw[(size_t)m * 64 + tid];
    __syncthreads();
    for (int c = tid; c < Csz; c += 256) {
        float acc = 0.0f;
#pragma unroll 8
        for (int d = 0; d < 64; ++d) acc += hs[d] * w2[(size_t)d * Csz + c];
        float u = w0[c] + acc;
        float w = -log1pf(expf(-u)) - 0.5f;        // -softplus(-u) - 0.5
        decay[(size_t)m * Csz + c] = expf(-expf(w));
    }
}

__global__ void lora_a_kernel(const float* __restrict__ Ha, const float* __restrict__ a2,
                              const float* __restrict__ a0, float* __restrict__ aOut) {
    int m = blockIdx.x, tid = threadIdx.x;
    __shared__ float hs[64];
    if (tid < 64) hs[tid] = Ha[(size_t)m * 64 + tid];
    __syncthreads();
    for (int c = tid; c < Csz; c += 256) {
        float acc = 0.0f;
#pragma unroll 8
        for (int d = 0; d < 64; ++d) acc += hs[d] * a2[(size_t)d * Csz + c];
        aOut[(size_t)m * Csz + c] = sigf(a0[c] + acc);
    }
}

__global__ void lora_v_kernel(const float* __restrict__ Hv, const float* __restrict__ v2,
                              const float* __restrict__ v0, float* __restrict__ vio,
                              const float* __restrict__ vfirst) {
    int m = blockIdx.x, tid = threadIdx.x;
    __shared__ float hs[64];
    if (tid < 64) hs[tid] = Hv[(size_t)m * 64 + tid];
    __syncthreads();
    for (int c = tid; c < Csz; c += 256) {
        float acc = 0.0f;
#pragma unroll 8
        for (int d = 0; d < 64; ++d) acc += hs[d] * v2[(size_t)d * Csz + c];
        float s = sigf(v0[c] + acc);
        size_t i = (size_t)m * Csz + c;
        float v = vio[i];
        vio[i] = v + (vfirst[i] - v) * s;
    }
}

__global__ void lora_g_kernel(const float* __restrict__ Hg, const float* __restrict__ g2,
                              float* __restrict__ gOut) {
    int m = blockIdx.x, tid = threadIdx.x;
    __shared__ float hs[160];
    if (tid < 160) hs[tid] = Hg[(size_t)m * 160 + tid];
    __syncthreads();
    for (int c = tid; c < Csz; c += 256) {
        float acc = 0.0f;
#pragma unroll 8
        for (int d = 0; d < 160; ++d) acc += hs[d] * g2[(size_t)d * Csz + c];
        gOut[(size_t)m * Csz + c] = acc;
    }
}

// ------------------------------------------ per-head kk normalization (wave32)
__global__ void kk_kernel(float* __restrict__ k,            // in: k, out: updated k
                          const float* __restrict__ a,
                          const float* __restrict__ k_k, const float* __restrict__ k_a,
                          float* __restrict__ aaOut, float* __restrict__ bbOut) {
    int bth = blockIdx.x;              // over B*T*H
    int lane = threadIdx.x;            // 0..31 (one wave)
    int h = bth & (Hn - 1);
    int bt = bth >> 4;
    size_t base = (size_t)bt * Csz + h * Nn;
    int c0 = h * Nn + lane, c1 = c0 + 32;
    float k0 = k[base + lane], k1 = k[base + lane + 32];
    float kk0 = k0 * k_k[c0], kk1 = k1 * k_k[c1];
    float ss = wave_sum32(kk0 * kk0 + kk1 * kk1);
    float inv = 1.0f / fmaxf(sqrtf(ss), 1e-12f);
    float kn0 = kk0 * inv, kn1 = kk1 * inv;
    float a0 = a[base + lane], a1 = a[base + lane + 32];
    aaOut[base + lane] = -kn0;          aaOut[base + lane + 32] = -kn1;
    bbOut[base + lane] = kn0 * a0;      bbOut[base + lane + 32] = kn1 * a1;
    k[base + lane]      = k0 * (1.0f + (a0 - 1.0f) * k_a[c0]);
    k[base + lane + 32] = k1 * (1.0f + (a1 - 1.0f) * k_a[c1]);
}

// --------------------------------------------------- sequential RWKV-7 scan
// one block per (b,h); thread i owns state row S[i][0..63] in registers
__global__ void scan_kernel(const float* __restrict__ r, const float* __restrict__ wdec,
                            const float* __restrict__ k, const float* __restrict__ v,
                            const float* __restrict__ aa, const float* __restrict__ bb,
                            float* __restrict__ y, float* __restrict__ saOut) {
    int bh = blockIdx.x;
    int b = bh / Hn, h = bh % Hn;
    int i = threadIdx.x;  // 0..63
    float S[Nn];
#pragma unroll
    for (int j = 0; j < Nn; ++j) S[j] = 0.0f;
    __shared__ float lr[Nn], lw[Nn], lk[Nn], la[Nn], lb[Nn];
    size_t base = (size_t)b * Tsz * Csz + h * Nn;
    for (int t = 0; t < Tsz; ++t) {
        size_t o = base + (size_t)t * Csz;
        lr[i] = r[o + i]; lw[i] = wdec[o + i]; lk[i] = k[o + i];
        la[i] = aa[o + i]; lb[i] = bb[o + i];
        float vi = v[o + i];
        __syncthreads();
        float sa = 0.0f;
#pragma unroll
        for (int j = 0; j < Nn; ++j) sa += S[j] * la[j];
        float yv = 0.0f;
#pragma unroll
        for (int j = 0; j < Nn; ++j) {
            float s = S[j] * lw[j] + sa * lb[j] + vi * lk[j];
            S[j] = s;
            yv += s * lr[j];
        }
        y[o + i] = yv;
        saOut[o + i] = sa;
        __syncthreads();
    }
}

// ----------------------------------------------- LayerNorm over C (state_rep)
__global__ void ln_kernel(const float* __restrict__ x, const float* __restrict__ w,
                          const float* __restrict__ bres, float* __restrict__ out) {
    int m = blockIdx.x, tid = threadIdx.x;
    __shared__ float rs[256], rq[256];
    float s = 0.0f, q = 0.0f;
    for (int c = tid; c < Csz; c += 256) {
        float v = x[(size_t)m * Csz + c];
        s += v; q += v * v;
    }
    rs[tid] = s; rq[tid] = q;
    __syncthreads();
    for (int off = 128; off > 0; off >>= 1) {
        if (tid < off) { rs[tid] += rs[tid + off]; rq[tid] += rq[tid + off]; }
        __syncthreads();
    }
    float mu = rs[0] / (float)Csz;
    float var = rq[0] / (float)Csz - mu * mu;
    float inv = rsqrtf(var + 1e-5f);
    for (int c = tid; c < Csz; c += 256) {
        size_t idx = (size_t)m * Csz + c;
        out[idx] = (x[idx] - mu) * inv * w[c] + bres[c];
    }
}

// ------------------- per-head GroupNorm + bonus + g gating -> bf16 GEMM input
__global__ void posthead_kernel(const float* __restrict__ xatt, const float* __restrict__ r,
                                const float* __restrict__ k, const float* __restrict__ v,
                                const float* __restrict__ rk, const float* __restrict__ gnw,
                                const float* __restrict__ gnb, const float* __restrict__ g,
                                bf16* __restrict__ xgf) {
    int bth = blockIdx.x;
    int lane = threadIdx.x;  // one wave
    int h = bth & (Hn - 1);
    int bt = bth >> 4;
    size_t base = (size_t)bt * Csz + h * Nn;
    int c0 = h * Nn + lane, c1 = c0 + 32;
    float x0 = xatt[base + lane], x1 = xatt[base + lane + 32];
    float s  = wave_sum32(x0 + x1);
    float sq = wave_sum32(x0 * x0 + x1 * x1);
    float mu = s * (1.0f / 64.0f);
    float var = sq * (1.0f / 64.0f) - mu * mu;
    float inv = rsqrtf(var + 0.00064f);
    float r0 = r[base + lane], r1 = r[base + lane + 32];
    float k0 = k[base + lane], k1 = k[base + lane + 32];
    float bd = wave_sum32(r0 * k0 * rk[c0] + r1 * k1 * rk[c1]);
    float xo0 = (x0 - mu) * inv * gnw[c0] + gnb[c0] + bd * v[base + lane];
    float xo1 = (x1 - mu) * inv * gnw[c1] + gnb[c1] + bd * v[base + lane + 32];
    xgf[base + lane]      = (bf16)(xo0 * g[base + lane]);
    xgf[base + lane + 32] = (bf16)(xo1 * g[base + lane + 32]);
}

__global__ void copy_f32_kernel(const float* __restrict__ src, float* __restrict__ dst, int n) {
    int i = blockIdx.x * blockDim.x + threadIdx.x;
    if (i < n) dst[i] = src[i];
}

// =============================================================================
extern "C" void kernel_launch(void* const* d_in, const int* in_sizes, int n_in,
                              void* d_out, int out_size, void* d_ws, size_t ws_size,
                              hipStream_t stream) {
    const float* x       = (const float*)d_in[0];
    const float* v_first = (const float*)d_in[1];
    const float* x_r = (const float*)d_in[2];
    const float* x_w = (const float*)d_in[3];
    const float* x_k = (const float*)d_in[4];
    const float* x_v = (const float*)d_in[5];
    const float* x_a = (const float*)d_in[6];
    const float* x_g = (const float*)d_in[7];
    const float* w0 = (const float*)d_in[8];
    const float* w1 = (const float*)d_in[9];
    const float* w2 = (const float*)d_in[10];
    const float* a0 = (const float*)d_in[11];
    const float* a1 = (const float*)d_in[12];
    const float* a2 = (const float*)d_in[13];
    const float* v0 = (const float*)d_in[14];
    const float* v1 = (const float*)d_in[15];
    const float* v2 = (const float*)d_in[16];
    const float* g1 = (const float*)d_in[17];
    const float* g2 = (const float*)d_in[18];
    const float* k_k = (const float*)d_in[19];
    const float* k_a = (const float*)d_in[20];
    const float* r_k = (const float*)d_in[21];
    const float* Wr = (const float*)d_in[22];
    const float* Wk = (const float*)d_in[23];
    const float* Wv = (const float*)d_in[24];
    const float* Wo = (const float*)d_in[25];
    const float* gn_w = (const float*)d_in[26];
    const float* gn_b = (const float*)d_in[27];
    const float* ln_w = (const float*)d_in[28];
    const float* ln_b = (const float*)d_in[29];

    float* out = (float*)d_out;                 // [MC] out | [MC] v_first | [MC] state_rep

    // ---- workspace bump allocator
    char* p = (char*)d_ws;
    auto alloc = [&](size_t bytes) -> void* {
        void* r = (void*)p;
        p += (bytes + 255) & ~(size_t)255;
        return r;
    };
    bf16* xr_b = (bf16*)alloc(MC * 2); bf16* xw_b = (bf16*)alloc(MC * 2);
    bf16* xk_b = (bf16*)alloc(MC * 2); bf16* xv_b = (bf16*)alloc(MC * 2);
    bf16* xa_b = (bf16*)alloc(MC * 2); bf16* xg_b = (bf16*)alloc(MC * 2);
    bf16* xgf  = (bf16*)alloc(MC * 2);
    bf16* Wr_b = (bf16*)alloc(CC * 2); bf16* Wk_b = (bf16*)alloc(CC * 2);
    bf16* Wv_b = (bf16*)alloc(CC * 2); bf16* Wo_b = (bf16*)alloc(CC * 2);
    float* rbuf  = (float*)alloc(MC * 4); float* kbuf = (float*)alloc(MC * 4);
    float* vbuf  = (float*)alloc(MC * 4); float* dec  = (float*)alloc(MC * 4);
    float* abuf  = (float*)alloc(MC * 4); float* gbuf = (float*)alloc(MC * 4);
    float* aabuf = (float*)alloc(MC * 4); float* bbbuf = (float*)alloc(MC * 4);
    float* xatt  = (float*)alloc(MC * 4); float* sabuf = (float*)alloc(MC * 4);
    float* hw = (float*)alloc((size_t)Msz * 64 * 4);
    float* ha = (float*)alloc((size_t)Msz * 64 * 4);
    float* hv = (float*)alloc((size_t)Msz * 64 * 4);
    float* hg = (float*)alloc((size_t)Msz * 160 * 4);

    // 1) weights -> bf16
    int nCC = (int)CC;
    cvt_bf16_kernel<<<(nCC + 255) / 256, 256, 0, stream>>>(Wr, Wr_b, nCC);
    cvt_bf16_kernel<<<(nCC + 255) / 256, 256, 0, stream>>>(Wk, Wk_b, nCC);
    cvt_bf16_kernel<<<(nCC + 255) / 256, 256, 0, stream>>>(Wv, Wv_b, nCC);
    cvt_bf16_kernel<<<(nCC + 255) / 256, 256, 0, stream>>>(Wo, Wo_b, nCC);

    // 2) token shift + mixes
    prep_mix_kernel<<<(int)(MC / 256), 256, 0, stream>>>(x, x_r, x_w, x_k, x_v, x_a, x_g,
                                                         xr_b, xw_b, xk_b, xv_b, xa_b, xg_b);

    // 3) big WMMA GEMMs: r, k, v
    //    grid: x = N/64 (block-wide N strip), y = M/128 (8 waves x 16 rows)
    dim3 gg(Csz / 64, Msz / 128);
    gemm_bf16_wmma_kernel<<<gg, 256, 0, stream>>>(xr_b, Wr_b, rbuf, Msz, Csz, Csz);
    gemm_bf16_wmma_kernel<<<gg, 256, 0, stream>>>(xk_b, Wk_b, kbuf, Msz, Csz, Csz);
    gemm_bf16_wmma_kernel<<<gg, 256, 0, stream>>>(xv_b, Wv_b, vbuf, Msz, Csz, Csz);

    // 4) LoRA stage 1
    lora_stage1_kernel<<<Msz, 64, 0, stream>>>(xw_b, w1, hw, Csz, 64, 1);   // tanh
    lora_stage1_kernel<<<Msz, 64, 0, stream>>>(xa_b, a1, ha, Csz, 64, 0);
    lora_stage1_kernel<<<Msz, 64, 0, stream>>>(xv_b, v1, hv, Csz, 64, 0);
    lora_stage1_kernel<<<Msz, 160, 0, stream>>>(xg_b, g1, hg, Csz, 160, 2); // sigmoid

    // 5) LoRA stage 2 / epilogues
    lora_w_kernel<<<Msz, 256, 0, stream>>>(hw, w2, w0, dec);
    lora_a_kernel<<<Msz, 256, 0, stream>>>(ha, a2, a0, abuf);
    lora_v_kernel<<<Msz, 256, 0, stream>>>(hv, v2, v0, vbuf, v_first);
    lora_g_kernel<<<Msz, 256, 0, stream>>>(hg, g2, gbuf);

    // 6) kk normalization; updates k in place, produces aa/bb
    kk_kernel<<<Msz * Hn, 32, 0, stream>>>(kbuf, abuf, k_k, k_a, aabuf, bbbuf);

    // 7) sequential delta-rule scan
    scan_kernel<<<Bsz * Hn, Nn, 0, stream>>>(rbuf, dec, kbuf, vbuf, aabuf, bbbuf, xatt, sabuf);

    // 8) state_rep = LayerNorm(sa_out) -> third output slab
    ln_kernel<<<Msz, 256, 0, stream>>>(sabuf, ln_w, ln_b, out + 2 * MC);

    // 9) per-head GroupNorm + bonus, gate with g -> bf16 input for Wo
    posthead_kernel<<<Msz * Hn, 32, 0, stream>>>(xatt, rbuf, kbuf, vbuf, r_k, gn_w, gn_b,
                                                 gbuf, xgf);

    // 10) out = (x_att * g) @ Wo^T  (WMMA) -> first output slab
    gemm_bf16_wmma_kernel<<<gg, 256, 0, stream>>>(xgf, Wo_b, out, Msz, Csz, Csz);

    // 11) v_first passthrough -> second output slab
    copy_f32_kernel<<<(int)(MC / 256), 256, 0, stream>>>(v_first, out + MC, (int)MC);
}
